// FusedTriangleMultiplicativeUpdate_24842090840416
// MI455X (gfx1250) — matrix operand order repaired and checked by reading, not compiled, not run on hardware
//
#include <hip/hip_runtime.h>
#include <hip/hip_bf16.h>

// ---------------------------------------------------------------------------
// FusedTriangleMultiplicativeUpdate for MI455X (gfx1250, wave32, WMMA)
//   N = 768, C_Z = C_H = 128, B = 1, fp32 in/out, bf16 tensor-core path.
// ---------------------------------------------------------------------------

typedef __attribute__((ext_vector_type(16))) __bf16 v16bf;
typedef __attribute__((ext_vector_type(8)))  __bf16 v8bf;
typedef __attribute__((ext_vector_type(8)))  float  v8f;
typedef __attribute__((ext_vector_type(4)))  float  v4f;

#define NDIM 768
#define CZ   128
#define CH   128
#define NN   (768*768)
#define LNEPS 1e-5f

// workspace layout (bytes)
//   [0, 196608)                      bf16 weights: ab_p(256x128) ab_g(256x128) g(128x128) z(128x128)
//   [WS_A, +NN*128*2)                a planes  [c][i][k] bf16
//   [WS_B, +NN*128*2)                b planes  [c][j][k] bf16
//   [WS_G, +NN*128*2)                gate      [i][j][c] bf16
static const size_t WS_A = 196608;
static const size_t WS_B = WS_A + (size_t)NN * CH * 2;
static const size_t WS_G = WS_B + (size_t)NN * CH * 2;

// ---------------------------------------------------------------------------
// Kernel 0: fp32 weights -> bf16 packed block in workspace
// ---------------------------------------------------------------------------
__global__ __launch_bounds__(256)
void k0_prep_weights(const float* __restrict__ pw, const float* __restrict__ gw,
                     const float* __restrict__ gatew, const float* __restrict__ zw,
                     __bf16* __restrict__ out)
{
    int idx = blockIdx.x * 256 + threadIdx.x;   // 0 .. 98303
    float v;
    if      (idx < 32768) v = pw[idx];
    else if (idx < 65536) v = gw[idx - 32768];
    else if (idx < 81920) v = gatew[idx - 65536];
    else                  v = zw[idx - 81920];
    out[idx] = (__bf16)v;
}

// ---------------------------------------------------------------------------
// Kernel 1: per 16-position tile (row i, cols j0..j0+15):
//   LayerNorm(z) -> zn (bf16, LDS) -> WMMA projections:
//     ab = (zn@Wp^T + pb) * sigmoid(zn@Wg^T + gb) * mask    (256 ch -> a|b planes)
//     gate = sigmoid(zn@Wgate^T + gateb)                    (128 ch -> gate buf)
// ---------------------------------------------------------------------------
__global__ __launch_bounds__(128)
void k1_ln_proj(const float* __restrict__ z, const float* __restrict__ mask,
                const float* __restrict__ niw, const float* __restrict__ nib,
                const __bf16* __restrict__ wbf,
                const float* __restrict__ pb, const float* __restrict__ gb,
                const float* __restrict__ gateb,
                __bf16* __restrict__ aPl, __bf16* __restrict__ bPl,
                __bf16* __restrict__ gPl)
{
    __shared__ __attribute__((aligned(16))) float  zraw[16 * 128];
    __shared__ float  red[16 * 8];
    __shared__ float  mu[16], rs[16], mtile[16];
    __shared__ __attribute__((aligned(16))) __bf16 znb[16 * 136];   // pad 136: conflict-free b128
    __shared__ __attribute__((aligned(16))) __bf16 gtile[16 * 128];

    const int t  = threadIdx.x;
    const int i  = blockIdx.y;
    const int j0 = blockIdx.x * 16;
    const size_t base = ((size_t)i * NDIM + j0) * CZ;

    // ---- load z tile (16x128 f32, 8KB, coalesced) ----
    #pragma unroll
    for (int it = 0; it < 4; ++it) {
        int f = it * 512 + t * 4;
        *(v4f*)&zraw[f] = *(const v4f*)&z[base + f];
    }
    if (t < 16) mtile[t] = mask[(size_t)i * NDIM + j0 + t];
    __syncthreads();

    // ---- LayerNorm over 128 channels (8 threads per row) ----
    const int r = t >> 3, s = t & 7;
    {
        float acc = 0.f;
        #pragma unroll
        for (int c = 0; c < 16; ++c) acc += zraw[r * 128 + s * 16 + c];
        red[r * 8 + s] = acc;
    }
    __syncthreads();
    if (t < 16) {
        float m = 0.f;
        #pragma unroll
        for (int q = 0; q < 8; ++q) m += red[t * 8 + q];
        mu[t] = m * (1.f / 128.f);
    }
    __syncthreads();
    {
        float m = mu[r], acc = 0.f;
        #pragma unroll
        for (int c = 0; c < 16; ++c) {
            float d = zraw[r * 128 + s * 16 + c] - m;
            acc += d * d;
        }
        red[r * 8 + s] = acc;
    }
    __syncthreads();
    if (t < 16) {
        float v = 0.f;
        #pragma unroll
        for (int q = 0; q < 8; ++q) v += red[t * 8 + q];
        rs[t] = rsqrtf(v * (1.f / 128.f) + LNEPS);
    }
    __syncthreads();
    {
        float m = mu[r], k = rs[r];
        #pragma unroll
        for (int c0 = 0; c0 < 16; ++c0) {
            int c = s * 16 + c0;
            znb[r * 136 + c] = (__bf16)((zraw[r * 128 + c] - m) * k * niw[c] + nib[c]);
        }
    }
    __syncthreads();

    // ---- WMMA projections ----
    const int lane = t & 31, wave = t >> 5;
    const int half = lane >> 4, l16 = lane & 15;

    // A fragments (zn tile) loaded once, reused by every output chunk.
    v16bf afrag[4];
    #pragma unroll
    for (int kc = 0; kc < 4; ++kc) {
        const __bf16* p = &znb[l16 * 136 + kc * 32 + half * 16];
        v8bf lo = *(const v8bf*)p;
        v8bf hi = *(const v8bf*)(p + 8);
        afrag[kc] = __builtin_shufflevector(lo, hi, 0,1,2,3,4,5,6,7,8,9,10,11,12,13,14,15);
    }

    const __bf16* Wp    = wbf;          // 256x128
    const __bf16* Wg    = wbf + 32768;  // 256x128
    const __bf16* Wgate = wbf + 65536;  // 128x128

    // ab: 16 chunks of 16 out-channels; each wave takes 4 (chunk = wave + 4*cc)
    for (int cc = 0; cc < 4; ++cc) {
        int dBase = (wave + cc * 4) * 16;
        v8f accP = {}, accG = {};
        #pragma unroll
        for (int kc = 0; kc < 4; ++kc) {
            v16bf bp = *(const v16bf*)&Wp[(size_t)(dBase + l16) * CZ + kc * 32 + half * 16];
            v16bf bg = *(const v16bf*)&Wg[(size_t)(dBase + l16) * CZ + kc * 32 + half * 16];
            accP = __builtin_amdgcn_wmma_f32_16x16x32_bf16(false, afrag[kc], false, bp,
                                                           (short)0, accP, false, false);
            accG = __builtin_amdgcn_wmma_f32_16x16x32_bf16(false, afrag[kc], false, bg,
                                                           (short)0, accG, false, false);
        }
        int d = dBase + l16;
        float biasP = pb[d], biasG = gb[d];
        v8bf outv;
        #pragma unroll
        for (int v = 0; v < 8; ++v) {
            int m = half * 8 + v;               // position within tile
            float pv = accP[v] + biasP;
            float gv = accG[v] + biasG;
            float sg = 1.f / (1.f + __expf(-gv));
            outv[v] = (__bf16)(pv * sg * mtile[m]);
        }
        __bf16* plane = (d < 128) ? aPl : bPl;  // uniform per chunk
        int dd = d & 127;
        *(v8bf*)&plane[(size_t)dd * NN + (size_t)i * NDIM + j0 + half * 8] = outv;
    }

    // gate: 8 chunks; each wave takes 2
    for (int cc = 0; cc < 2; ++cc) {
        int dBase = (wave + cc * 4) * 16;
        v8f acc = {};
        #pragma unroll
        for (int kc = 0; kc < 4; ++kc) {
            v16bf bw = *(const v16bf*)&Wgate[(size_t)(dBase + l16) * CZ + kc * 32 + half * 16];
            acc = __builtin_amdgcn_wmma_f32_16x16x32_bf16(false, afrag[kc], false, bw,
                                                          (short)0, acc, false, false);
        }
        int d = dBase + l16;
        float bias = gateb[d];
        #pragma unroll
        for (int v = 0; v < 8; ++v) {
            int m = half * 8 + v;
            gtile[m * 128 + d] = (__bf16)(1.f / (1.f + __expf(-(acc[v] + bias))));
        }
    }
    __syncthreads();

    // coalesced 4KB gate store, position-major [i][j][c]
    #pragma unroll
    for (int it = 0; it < 2; ++it) {
        int f = it * 1024 + t * 8;
        *(v8bf*)&gPl[base + f] = *(const v8bf*)&gtile[f];
    }
}

// ---------------------------------------------------------------------------
// Kernel 2: x[i,j,c] = sum_k a[i,k,c]*b[j,k,c]
//   128 independent 768x768x768 NT GEMMs. Block = 256 thr (8 waves) computes a
//   16x16 (i,j) tile for ALL 128 channels (wave w: channels 16w..16w+15, two
//   accumulator chains in flight). f32 results staged in padded LDS, then a
//   fully coalesced [i][j][c] write to d_out.
// ---------------------------------------------------------------------------
__global__ __launch_bounds__(256)
void k2_triangle(const __bf16* __restrict__ aPl, const __bf16* __restrict__ bPl,
                 float* __restrict__ x)
{
    extern __shared__ float xs[];               // 256 * 132 floats = 135168 B
    const int t    = threadIdx.x;
    const int lane = t & 31, wave = t >> 5;
    const int half = lane >> 4, l16 = lane & 15;
    const int i0 = blockIdx.y * 16, j0 = blockIdx.x * 16;

    for (int p = 0; p < 8; ++p) {
        const int c0 = wave * 16 + p * 2, c1 = c0 + 1;
        const __bf16* a0 = aPl + (size_t)c0 * NN + (size_t)(i0 + l16) * NDIM;
        const __bf16* a1 = aPl + (size_t)c1 * NN + (size_t)(i0 + l16) * NDIM;
        const __bf16* b0 = bPl + (size_t)c0 * NN + (size_t)(j0 + l16) * NDIM;
        const __bf16* b1 = bPl + (size_t)c1 * NN + (size_t)(j0 + l16) * NDIM;
        v8f acc0 = {}, acc1 = {};

        for (int kb = 0; kb < NDIM; kb += 32) {
            const int ka = kb + half * 16;      // A: two 16B per lane
            const int kc = kb + half * 16;      // B: one 32B per lane
            v8bf a0lo = *(const v8bf*)(a0 + ka);
            v8bf a0hi = *(const v8bf*)(a0 + ka + 8);
            v8bf a1lo = *(const v8bf*)(a1 + ka);
            v8bf a1hi = *(const v8bf*)(a1 + ka + 8);
            v16bf A0 = __builtin_shufflevector(a0lo, a0hi, 0,1,2,3,4,5,6,7,8,9,10,11,12,13,14,15);
            v16bf A1 = __builtin_shufflevector(a1lo, a1hi, 0,1,2,3,4,5,6,7,8,9,10,11,12,13,14,15);
            v16bf B0 = *(const v16bf*)(b0 + kc);
            v16bf B1 = *(const v16bf*)(b1 + kc);
            if (kb + 32 < NDIM) {               // prefetch next k-step (L2-hot planes)
                __builtin_prefetch(a0 + ka + 32, 0, 3);
                __builtin_prefetch(b0 + kc + 32, 0, 3);
            }
            acc0 = __builtin_amdgcn_wmma_f32_16x16x32_bf16(false, A0, false, B0,
                                                           (short)0, acc0, false, false);
            acc1 = __builtin_amdgcn_wmma_f32_16x16x32_bf16(false, A1, false, B1,
                                                           (short)0, acc1, false, false);
        }

        // stage C tiles: lane holds col j=l16, rows i = 8*half+v; pack (c0,c1)
        #pragma unroll
        for (int v = 0; v < 8; ++v) {
            int il = half * 8 + v;
            float2 pr = make_float2(acc0[v], acc1[v]);
            *(float2*)&xs[(il * 16 + l16) * 132 + c0] = pr;
        }
    }
    __syncthreads();

    // write out: per i-row, a contiguous 16x128 f32 = 8KB run
    for (int il = 0; il < 16; ++il) {
        size_t gbase = ((size_t)(i0 + il) * NDIM + j0) * CH;
        #pragma unroll
        for (int it = 0; it < 2; ++it) {
            int f  = it * 1024 + t * 4;
            int jl = f >> 7, c = f & 127;
            *(v4f*)&x[gbase + f] = *(const v4f*)&xs[(il * 16 + jl) * 132 + c];
        }
    }
}

// ---------------------------------------------------------------------------
// Kernel 3: out = (LN(x) @ z_weight^T + z_bias) * gate   (in-place on d_out)
// ---------------------------------------------------------------------------
__global__ __launch_bounds__(128)
void k3_out(const float* __restrict__ now, const float* __restrict__ nob,
            const __bf16* __restrict__ Wz, const float* __restrict__ zb,
            const __bf16* __restrict__ gPl, float* __restrict__ xo)
{
    __shared__ __attribute__((aligned(16))) float  xraw[16 * 128];
    __shared__ float  red[16 * 8];
    __shared__ float  mu[16], rs[16];
    __shared__ __attribute__((aligned(16))) __bf16 xnb[16 * 136];
    __shared__ __attribute__((aligned(16))) __bf16 gt[16 * 128];
    __shared__ __attribute__((aligned(16))) float  ot[16 * 128];

    const int t  = threadIdx.x;
    const int i  = blockIdx.y;
    const int j0 = blockIdx.x * 16;
    const size_t base = ((size_t)i * NDIM + j0) * CH;

    #pragma unroll
    for (int it = 0; it < 4; ++it) {
        int f = it * 512 + t * 4;
        *(v4f*)&xraw[f] = *(const v4f*)&xo[base + f];
    }
    #pragma unroll
    for (int it = 0; it < 2; ++it) {            // gate tile, coalesced bf16
        int f = it * 1024 + t * 8;
        *(v8bf*)&gt[f] = *(const v8bf*)&gPl[base + f];
    }
    __syncthreads();

    const int r = t >> 3, s = t & 7;
    {
        float acc = 0.f;
        #pragma unroll
        for (int c = 0; c < 16; ++c) acc += xraw[r * 128 + s * 16 + c];
        red[r * 8 + s] = acc;
    }
    __syncthreads();
    if (t < 16) {
        float m = 0.f;
        #pragma unroll
        for (int q = 0; q < 8; ++q) m += red[t * 8 + q];
        mu[t] = m * (1.f / 128.f);
    }
    __syncthreads();
    {
        float m = mu[r], acc = 0.f;
        #pragma unroll
        for (int c = 0; c < 16; ++c) {
            float d = xraw[r * 128 + s * 16 + c] - m;
            acc += d * d;
        }
        red[r * 8 + s] = acc;
    }
    __syncthreads();
    if (t < 16) {
        float v = 0.f;
        #pragma unroll
        for (int q = 0; q < 8; ++q) v += red[t * 8 + q];
        rs[t] = rsqrtf(v * (1.f / 128.f) + LNEPS);
    }
    __syncthreads();
    {
        float m = mu[r], k = rs[r];
        #pragma unroll
        for (int c0 = 0; c0 < 16; ++c0) {
            int c = s * 16 + c0;
            xnb[r * 136 + c] = (__bf16)((xraw[r * 128 + c] - m) * k * now[c] + nob[c]);
        }
    }
    __syncthreads();

    const int lane = t & 31, wave = t >> 5;
    const int half = lane >> 4, l16 = lane & 15;

    v16bf afrag[4];
    #pragma unroll
    for (int kc = 0; kc < 4; ++kc) {
        const __bf16* p = &xnb[l16 * 136 + kc * 32 + half * 16];
        v8bf lo = *(const v8bf*)p;
        v8bf hi = *(const v8bf*)(p + 8);
        afrag[kc] = __builtin_shufflevector(lo, hi, 0,1,2,3,4,5,6,7,8,9,10,11,12,13,14,15);
    }

    for (int cc = 0; cc < 2; ++cc) {
        int dBase = (wave + cc * 4) * 16;
        v8f acc = {};
        #pragma unroll
        for (int kc = 0; kc < 4; ++kc) {
            v16bf bw = *(const v16bf*)&Wz[(size_t)(dBase + l16) * CH + kc * 32 + half * 16];
            acc = __builtin_amdgcn_wmma_f32_16x16x32_bf16(false, afrag[kc], false, bw,
                                                          (short)0, acc, false, false);
        }
        int d = dBase + l16;
        float bias = zb[d];
        #pragma unroll
        for (int v = 0; v < 8; ++v) {
            int m = half * 8 + v;
            ot[m * 128 + d] = (acc[v] + bias) * (float)gt[m * 128 + d];
        }
    }
    __syncthreads();

    #pragma unroll
    for (int it = 0; it < 4; ++it) {
        int f = it * 512 + t * 4;
        *(v4f*)&xo[base + f] = *(const v4f*)&ot[f];
    }
}

// ---------------------------------------------------------------------------
extern "C" void kernel_launch(void* const* d_in, const int* in_sizes, int n_in,
                              void* d_out, int out_size, void* d_ws, size_t ws_size,
                              hipStream_t stream)
{
    (void)in_sizes; (void)n_in; (void)out_size; (void)ws_size;
    const float* z     = (const float*)d_in[0];
    const float* mask  = (const float*)d_in[1];
    const float* niw   = (const float*)d_in[2];
    const float* nib   = (const float*)d_in[3];
    const float* pw    = (const float*)d_in[4];
    const float* pbias = (const float*)d_in[5];
    const float* gw    = (const float*)d_in[6];
    const float* gbias = (const float*)d_in[7];
    const float* gatew = (const float*)d_in[8];
    const float* gateb = (const float*)d_in[9];
    const float* now   = (const float*)d_in[10];
    const float* nob   = (const float*)d_in[11];
    const float* zw    = (const float*)d_in[12];
    const float* zbias = (const float*)d_in[13];
    float* out = (float*)d_out;

    char* ws = (char*)d_ws;
    __bf16* wbf = (__bf16*)ws;
    __bf16* aPl = (__bf16*)(ws + WS_A);
    __bf16* bPl = (__bf16*)(ws + WS_B);
    __bf16* gPl = (__bf16*)(ws + WS_G);

    k0_prep_weights<<<384, 256, 0, stream>>>(pw, gw, gatew, zw, wbf);
    k1_ln_proj<<<dim3(48, 768), 128, 0, stream>>>(z, mask, niw, nib, wbf,
                                                  pbias, gbias, gateb, aPl, bPl, gPl);
    k2_triangle<<<dim3(48, 48), 256, 256 * 132 * sizeof(float), stream>>>(aPl, bPl, out);
    k3_out<<<dim3(48, 768), 128, 0, stream>>>(now, nob, wbf + 81920, zbias, gPl, out);
}
